// PosNegBalanceLoss_17987323036121
// MI455X (gfx1250) — compile-verified, matrix operand order checked by reading, not run on hardware
//
#include <hip/hip_runtime.h>
#include <hip/hip_bf16.h>
#include <math.h>

typedef __attribute__((ext_vector_type(16))) _Float16 v16h;
typedef __attribute__((ext_vector_type(8)))  _Float16 v8h;
typedef __attribute__((ext_vector_type(2)))  _Float16 v2h;
typedef __attribute__((ext_vector_type(8)))  float    v8f;
typedef __attribute__((ext_vector_type(4)))  unsigned tdm_u4;
typedef __attribute__((ext_vector_type(8)))  int      tdm_i8;
typedef __attribute__((ext_vector_type(4)))  int      tdm_i4;

#define NCLS   40
#define CP     48            // padded classes (3 WMMA tiles of 16)
#define TROWS  128           // rows per tile (4 waves x 32 rows)
#define RPAD   8
#define RSTR   (TROWS+RPAD)  // 136 halves: 272B row stride -> bank step 4, 16B aligned
#define NB1    512
#define NB3    512
#define NB5    512
#define NT35   320           // multiple of 40 and 32
#define HBINS  4096
#define PSTR   12

#if __has_builtin(__builtin_amdgcn_tensor_load_to_lds) && \
    __has_builtin(__builtin_amdgcn_s_wait_tensorcnt)
#define USE_TDM 1
#else
#define USE_TDM 0
#endif

__device__ __forceinline__ float bce_f(float x, float t) {
  float ax = fabsf(x);
  return fmaxf(x, 0.0f) - x * t + __logf(1.0f + __expf(-ax));
}
__device__ __forceinline__ float sigm(float x) { return 1.0f / (1.0f + __expf(-x)); }
__device__ __forceinline__ unsigned lds_off(const void* p) {
  return (unsigned)(unsigned long long)p;   // LDS offset lives in the low 32 bits
}

// ---------------------------------------------------------------- k0: zero u32 scratch
__global__ void k0_zero(unsigned* __restrict__ p, int n) {
  int i = blockIdx.x * blockDim.x + threadIdx.x;
  if (i < n) p[i] = 0u;
}

// ---------------------------------------------------------------- k1: column sums via WMMA (+TDM staging)
// partials[block][0..47] = bce col sums, [48..95] = target col sums
__global__ __launch_bounds__(128) void k1_colsum_wmma(
    const float* __restrict__ pred, const float* __restrict__ tgt,
    float* __restrict__ partials, int Brows)
{
#if USE_TDM
  __shared__ __align__(16) float sPredRaw[TROWS * NCLS];   // 20KB, filled by TDM
#endif
  __shared__ __align__(16) _Float16 sBce[CP * RSTR];       // 13KB
  __shared__ __align__(16) _Float16 sTgt[CP * RSTR];       // 13KB
  __shared__ float sRed[4 * 2 * CP];

  const int tid  = threadIdx.x;
  const int lane = tid & 31;
  const int w    = tid >> 5;           // wave 0..3
  const int col  = lane & 15;
  const int hs   = lane >> 4;

  // zero the 8 pad-class columns once (contiguous tail of [class][row] layout)
  for (int e = tid; e < 8 * RSTR; e += 128) {
    sBce[NCLS * RSTR + e] = (_Float16)0.0f;
    sTgt[NCLS * RSTR + e] = (_Float16)0.0f;
  }

  v16h ones;
#pragma unroll
  for (int i = 0; i < 16; ++i) ones[i] = (_Float16)1.0f;

  v8f accB[3], accT[3];
#pragma unroll
  for (int ct = 0; ct < 3; ++ct)
#pragma unroll
    for (int i = 0; i < 8; ++i) { accB[ct][i] = 0.0f; accT[ct][i] = 0.0f; }

  const long total  = (long)Brows * NCLS;
  const int  ntiles = (Brows + TROWS - 1) / TROWS;
  const int  TELEMS = TROWS * NCLS;    // 5120

  for (int tile = blockIdx.x; tile < ntiles; tile += gridDim.x) {
    const long base = (long)tile * TELEMS;
    __syncthreads();   // previous iteration's LDS consumers done

#if USE_TDM
    if (w == 0) {
      // D# for a contiguous copy: 2-D tensor, dim1 = 1, tile 5120 x 1, 4B elems
      long rem = total - base;
      unsigned td0 = (unsigned)(rem < (long)TELEMS ? rem : (long)TELEMS);
      unsigned long long ga = (unsigned long long)(const void*)(pred + base);
      tdm_u4 g0 = { 1u,                                   // count=1, user mode
                    lds_off(sPredRaw),                    // lds_addr
                    (unsigned)(ga & 0xffffffffu),
                    (unsigned)(((ga >> 32) & 0x1ffffffu) | (2u << 30)) };  // type=2
      tdm_i8 g1 = { (int)(2u << 16),                      // data_size=4B, mask=0
                    (int)((td0 & 0xffffu) << 16),         // tensor_dim0[15:0]
                    (int)((td0 >> 16) | (1u << 16)),      // dim0[31:16] | tensor_dim1=1
                    (int)((unsigned)TELEMS << 16),        // dim1 hi=0 | tile_dim0
                    (int)1u,                              // tile_dim1=1, tile_dim2=0
                    (int)(unsigned)TELEMS,                // tensor_dim0_stride lo
                    0, 0 };
      tdm_i4 gz = { 0, 0, 0, 0 };
#if defined(__clang_major__) && (__clang_major__ >= 23)
      tdm_i8 gz8 = { 0,0,0,0,0,0,0,0 };
      __builtin_amdgcn_tensor_load_to_lds(g0, g1, gz, gz, gz8, 0);
#else
      __builtin_amdgcn_tensor_load_to_lds(g0, g1, gz, gz, 0);
#endif
      __builtin_amdgcn_s_wait_tensorcnt(0);
    }
#endif
    __syncthreads();   // TDM data visible (wave0 waited before signaling)

    // convert: two consecutive rows per thread -> one b32 store per matrix
#pragma unroll
    for (int j = 0; j < (TELEMS / 2) / 128; ++j) {
      int  pf = tid + j * 128;
      int  c  = pf % NCLS;
      int  r0 = (pf / NCLS) * 2;
      long gi0 = base + (long)r0 * NCLS + c;
      bool v0 = gi0 < total, v1 = (gi0 + NCLS) < total;
      float t0 = v0 ? tgt[gi0] : 0.0f;
      float t1 = v1 ? tgt[gi0 + NCLS] : 0.0f;
#if USE_TDM
      float x0 = sPredRaw[r0 * NCLS + c];
      float x1 = sPredRaw[(r0 + 1) * NCLS + c];
#else
      float x0 = v0 ? pred[gi0] : 0.0f;
      float x1 = v1 ? pred[gi0 + NCLS] : 0.0f;
#endif
      float b0 = v0 ? bce_f(x0, t0) : 0.0f;
      float b1 = v1 ? bce_f(x1, t1) : 0.0f;
      v2h pb = { (_Float16)b0, (_Float16)b1 };
      v2h pt = { (_Float16)t0, (_Float16)t1 };
      *(v2h*)&sBce[c * RSTR + r0] = pb;
      *(v2h*)&sTgt[c * RSTR + r0] = pt;
    }
    __syncthreads();

    const int base_row = w * 32 + hs * 16;
#pragma unroll
    for (int ct = 0; ct < 3; ++ct) {
      const int cc = ct * 16 + col;
      v8h blo = *(const v8h*)&sBce[cc * RSTR + base_row];
      v8h bhi = *(const v8h*)&sBce[cc * RSTR + base_row + 8];
      v8h tlo = *(const v8h*)&sTgt[cc * RSTR + base_row];
      v8h thi = *(const v8h*)&sTgt[cc * RSTR + base_row + 8];
      v16h bb = __builtin_shufflevector(blo, bhi, 0,1,2,3,4,5,6,7,8,9,10,11,12,13,14,15);
      v16h bt = __builtin_shufflevector(tlo, thi, 0,1,2,3,4,5,6,7,8,9,10,11,12,13,14,15);
      accB[ct] = __builtin_amdgcn_wmma_f32_16x16x32_f16(
          false, ones, false, bb, (short)0, accB[ct], false, false);
      accT[ct] = __builtin_amdgcn_wmma_f32_16x16x32_f16(
          false, ones, false, bt, (short)0, accT[ct], false, false);
    }
  }
  __syncthreads();

  if (lane < 16) {
#pragma unroll
    for (int ct = 0; ct < 3; ++ct) {
      sRed[w * (2 * CP) + ct * 16 + lane]      = accB[ct][0];
      sRed[w * (2 * CP) + CP + ct * 16 + lane] = accT[ct][0];
    }
  }
  __syncthreads();
  if (tid < 2 * CP) {
    float s = 0.0f;
#pragma unroll
    for (int ww = 0; ww < 4; ++ww) s += sRed[ww * (2 * CP) + tid];
    partials[blockIdx.x * (2 * CP) + tid] = s;
  }
}

// ---------------------------------------------------------------- k2: per-class scalar stats
__global__ __launch_bounds__(128) void k2_stats(
    const float* __restrict__ partials, const float* __restrict__ hard_rand,
    const float* __restrict__ pos_prop, float* __restrict__ params,
    int* __restrict__ dropk, int Brows, int nb1)
{
  __shared__ float cl[CP], cpos[CP], lns[NCLS], mm[2];
  int tid = threadIdx.x;
  if (tid < 2 * CP) {
    float s = 0.0f;
    for (int b = 0; b < nb1; ++b) s += partials[b * (2 * CP) + tid];
    if (tid < CP) cl[tid] = s; else cpos[tid - CP] = s;
  }
  __syncthreads();
  if (tid < NCLS) lns[tid] = log10f(1.0f + cl[tid]);
  __syncthreads();
  if (tid == 0) {
    float mn = lns[0], mx = lns[0];
    for (int c = 1; c < NCLS; ++c) { mn = fminf(mn, lns[c]); mx = fmaxf(mx, lns[c]); }
    mm[0] = mn; mm[1] = mx;
  }
  __syncthreads();
  if (tid < NCLS) {
    float Bf      = (float)Brows;
    float norm    = 5.0f - 10.0f * (lns[tid] - mm[0]) / (mm[1] - mm[0]);
    float drate   = fmaxf(sigm(norm), 0.0f);
    float pos     = cpos[tid], neg = Bf - pos;
    float bal_pos = pos_prop[tid] * Bf, bal_neg = Bf - bal_pos;
    bool  pos_gt  = pos > bal_pos, neg_gt = neg > bal_neg;
    float balance = pos_gt ? bal_pos : (neg_gt ? bal_neg : 0.0f);
    float dropn   = floorf(pos_gt ? (pos - bal_pos) : (neg_gt ? (neg - bal_neg) : 0.0f));
    bool  hard    = hard_rand[tid] > drate;
    float maj_lab = pos_gt ? 1.0f : 0.0f;
    float min_lab = neg_gt ? 1.0f : 0.0f;
    float maj_cnt = pos_gt ? pos : neg;
    float min_cnt = neg_gt ? pos : neg;
    float w_hard  = balance / fmaxf(maj_cnt, 1.0f);
    float w_easy  = (Bf - balance) / fmaxf(min_cnt, 1.0f);
    float* p = params + tid * PSTR;
    p[0] = drate;  p[1] = hard ? 1.0f : 0.0f;
    p[2] = maj_lab; p[3] = min_lab;
    p[4] = w_hard; p[5] = w_easy;
    p[6] = (min_cnt > 0.0f) ? 1.0f : 0.0f;
    p[7] = dropn;
    p[8] = (!hard && dropn >= 1.0f) ? 1.0f : 0.0f;
    dropk[tid] = (int)dropn;
  }
}

// ---------------------------------------------------------------- k3: per-class histogram of g (majority only)
__global__ __launch_bounds__(NT35) void k3_hist(
    const float* __restrict__ pred, const float* __restrict__ tgt,
    const float* __restrict__ params, unsigned* __restrict__ hist, long total)
{
  long t0 = (long)blockIdx.x * NT35 + threadIdx.x;
  int  c  = (int)(t0 % NCLS);
  const float* p = params + c * PSTR;
  if (p[8] < 0.5f) return;
  float maj = p[2];
  long  S   = (long)gridDim.x * NT35;
  unsigned* h = hist + (long)c * HBINS;
  for (long i = t0; i < total; i += S) {
    __builtin_prefetch(pred + i + S, 0, 1);
    float t = tgt[i];
    if (t == maj) {
      float g   = fabsf(sigm(pred[i]) - t);
      int   bin = (int)(g * (float)HBINS);
      bin = bin > HBINS - 1 ? HBINS - 1 : (bin < 0 ? 0 : bin);
      atomicAdd(&h[bin], 1u);
    }
  }
}

// ---------------------------------------------------------------- k4: scan -> threshold bin + residual
__global__ __launch_bounds__(64) void k4_scan(
    const unsigned* __restrict__ hist, const float* __restrict__ params,
    const int* __restrict__ dropk, int* __restrict__ bstar, int* __restrict__ resid)
{
  int c = threadIdx.x;
  if (c >= NCLS) return;
  if (params[c * PSTR + 8] < 0.5f) { bstar[c] = -1; resid[c] = 0; return; }
  int k = dropk[c];
  const unsigned* h = hist + (long)c * HBINS;
  long cum = 0; int b = -1, r = 0;
  for (int i = 0; i < HBINS; ++i) {
    long nc = cum + (long)h[i];
    if (nc >= (long)k) { b = i; r = k - (int)cum; break; }
    cum = nc;
  }
  if (b < 0) { b = HBINS - 1; r = 0; }
  bstar[c] = b; resid[c] = r;
}

// ---------------------------------------------------------------- k5: weighted BCE partials
__global__ __launch_bounds__(NT35) void k5_loss(
    const float* __restrict__ pred, const float* __restrict__ tgt,
    const float* __restrict__ rnd, const float* __restrict__ params,
    const int* __restrict__ bstar, const int* __restrict__ resid,
    unsigned* __restrict__ ticket, float* __restrict__ lpart, long total)
{
  __shared__ float sr[NT35];
  long t0 = (long)blockIdx.x * NT35 + threadIdx.x;
  int  c  = (int)(t0 % NCLS);
  const float* p = params + c * PSTR;
  float drate = p[0];
  bool  hard  = p[1] > 0.5f;
  float maj   = p[2], mnl = p[3], w_hard = p[4], w_easy = p[5];
  bool  easy_app = p[6] > 0.5f;
  int      bs = bstar[c];
  unsigned rs = (unsigned)resid[c];
  long S = (long)gridDim.x * NT35;
  float acc = 0.0f;
  for (long i = t0; i < total; i += S) {
    __builtin_prefetch(pred + i + S, 0, 1);
    __builtin_prefetch(rnd  + i + S, 0, 1);
    float x = pred[i], t = tgt[i], r = rnd[i];
    float b = bce_f(x, t);
    float g = fabsf(sigm(x) - t);
    bool is_maj = (t == maj), is_min = (t == mnl);
    float wgt;
    if (hard) {
      wgt = is_maj ? w_hard : 1.0f;
    } else {
      wgt = (is_min && easy_app) ? w_easy : 1.0f;
      if (is_maj && bs >= 0) {
        int bin = (int)(g * (float)HBINS);
        bin = bin > HBINS - 1 ? HBINS - 1 : (bin < 0 ? 0 : bin);
        if (bin < bs) wgt = 0.0f;
        else if (bin == bs) { if (atomicAdd(&ticket[c], 1u) < rs) wgt = 0.0f; }
      }
    }
    if (g >= 0.8f && g < 1.000001f && r > drate) wgt = 0.0f;
    acc += wgt * b;
  }
  sr[threadIdx.x] = acc;
  __syncthreads();
  if (threadIdx.x < 160) sr[threadIdx.x] += sr[threadIdx.x + 160];
  __syncthreads();
  if (threadIdx.x <  80) sr[threadIdx.x] += sr[threadIdx.x +  80];
  __syncthreads();
  if (threadIdx.x <  40) sr[threadIdx.x] += sr[threadIdx.x +  40];
  __syncthreads();
  if (threadIdx.x == 0) {
    float s = 0.0f;
    for (int i = 0; i < 40; ++i) s += sr[i];
    lpart[blockIdx.x] = s;
  }
}

// ---------------------------------------------------------------- k6: final scalar
__global__ __launch_bounds__(256) void k6_final(
    const float* __restrict__ lpart, int n, float invN, float* __restrict__ out)
{
  __shared__ float sr[256];
  float s = 0.0f;
  for (int i = threadIdx.x; i < n; i += 256) s += lpart[i];
  sr[threadIdx.x] = s;
  __syncthreads();
  for (int st = 128; st > 0; st >>= 1) {
    if (threadIdx.x < st) sr[threadIdx.x] += sr[threadIdx.x + st];
    __syncthreads();
  }
  if (threadIdx.x == 0) out[0] = sr[0] * invN;
}

// ---------------------------------------------------------------- launcher
extern "C" void kernel_launch(void* const* d_in, const int* in_sizes, int n_in,
                              void* d_out, int out_size, void* d_ws, size_t ws_size,
                              hipStream_t stream) {
  const float* pred      = (const float*)d_in[0];
  const float* target    = (const float*)d_in[1];
  const float* rand_mat  = (const float*)d_in[2];
  const float* hard_rand = (const float*)d_in[3];
  const float* pos_prop  = (const float*)d_in[4];

  const int  Brows = in_sizes[0] / NCLS;
  const long total = (long)Brows * NCLS;

  float* wf       = (float*)d_ws;
  float* partials = wf;                          // NB1 * 96
  float* params   = partials + NB1 * (2 * CP);   // 40 * 12
  float* lpart    = params + NCLS * PSTR;        // NB5
  unsigned* hist  = (unsigned*)(lpart + NB5);    // 40 * 4096
  int* bstar      = (int*)(hist + (long)NCLS * HBINS);
  int* resid      = bstar + NCLS;
  unsigned* ticket= (unsigned*)(resid + NCLS);
  int* dropk      = (int*)(ticket + NCLS);

  const int nzero = NCLS * HBINS + 4 * NCLS;
  k0_zero<<<(nzero + 255) / 256, 256, 0, stream>>>(hist, nzero);

  k1_colsum_wmma<<<NB1, 128, 0, stream>>>(pred, target, partials, Brows);
  k2_stats<<<1, 128, 0, stream>>>(partials, hard_rand, pos_prop, params, dropk, Brows, NB1);
  k3_hist<<<NB3, NT35, 0, stream>>>(pred, target, params, hist, total);
  k4_scan<<<1, 64, 0, stream>>>(hist, params, dropk, bstar, resid);
  k5_loss<<<NB5, NT35, 0, stream>>>(pred, target, rand_mat, params, bstar, resid,
                                    ticket, lpart, total);
  k6_final<<<1, 256, 0, stream>>>(lpart, NB5, 1.0f / (float)total, (float*)d_out);
}
